// FullAttention_8796093022779
// MI455X (gfx1250) — matrix operand order repaired and checked by reading, not compiled, not run on hardware
//
#include <hip/hip_runtime.h>

typedef __attribute__((ext_vector_type(16))) __bf16 v16bf;
typedef __attribute__((ext_vector_type(8)))  float  v8f;

#define B_    4
#define L_    2048
#define H_    16
#define E_    64
#define BM    128      // query rows per workgroup (8 waves x 16)
#define BN    32       // keys per tile
#define NWAVE 8
#define KBSTR 72       // padded bf16 stride for K tile rows   (144B, 16B aligned)
#define VBSTR 40       // padded bf16 stride for V^T tile rows (80B, 16B aligned)
#define PSTR  24       // padded bf16 stride for P^T rows      (48B, 16B aligned)
#define KSTR  68       // fallback: padded f32 stride K rows
#define VSTR  36       // fallback: padded f32 stride V^T rows

// ---- raw IEEE maximumNumber (hardware already quiets NaNs; skip canonicalize) ----
__device__ __forceinline__ float fmax_raw(float a, float b) {
    float r;
    asm("v_max_num_f32 %0, %1, %2" : "=v"(r) : "v"(a), "v"(b));
    return r;
}

// ---- v_permlane16_b32 without the tied-old v_mov (EXEC all-ones, FI=0 -> old unused) ----
__device__ __forceinline__ float pl16x(float x, unsigned s0, unsigned s1) {
    float r;
    asm("v_permlane16_b32 %0, %1, %2, %3" : "=v"(r) : "v"(x), "s"(s0), "s"(s1));
    return r;
}

__device__ __forceinline__ float redmax16(float x) {
    x = fmax_raw(x, pl16x(x, 0x67452301u, 0xEFCDAB89u)); // xor 1
    x = fmax_raw(x, pl16x(x, 0x54761032u, 0xDCFE98BAu)); // xor 2
    x = fmax_raw(x, pl16x(x, 0x32107654u, 0xBA98FEDCu)); // xor 4
    x = fmax_raw(x, pl16x(x, 0xFEDCBA98u, 0x76543210u)); // xor 8
    return x;
}

__device__ __forceinline__ float redsum16(float x) {
    x += pl16x(x, 0x67452301u, 0xEFCDAB89u);
    x += pl16x(x, 0x54761032u, 0xDCFE98BAu);
    x += pl16x(x, 0x32107654u, 0xBA98FEDCu);
    x += pl16x(x, 0xFEDCBA98u, 0x76543210u);
    return x;
}

// ================= pre-pass: K f32 [b,s,h,e] -> bf16 [b,h,s,e] =================
__global__ __launch_bounds__(256)
void cvt_k_bf16(const float* __restrict__ K, __bf16* __restrict__ Kb)
{
    const size_t idx = ((size_t)blockIdx.x * 256 + threadIdx.x) * 8; // dest index
    const int e0 = (int)(idx & (E_ - 1));
    const size_t r  = idx >> 6;
    const int s  = (int)(r & (L_ - 1));
    const size_t r2 = r >> 11;
    const int h  = (int)(r2 & (H_ - 1));
    const int b  = (int)(r2 >> 4);
    const float* src = K + (((size_t)b * L_ + s) * H_ + h) * E_ + e0;
    const float4 a = *reinterpret_cast<const float4*>(src);
    const float4 c = *reinterpret_cast<const float4*>(src + 4);
    __align__(16) __bf16 o[8] = { (__bf16)a.x,(__bf16)a.y,(__bf16)a.z,(__bf16)a.w,
                                  (__bf16)c.x,(__bf16)c.y,(__bf16)c.z,(__bf16)c.w };
    *reinterpret_cast<uint4*>(Kb + idx) = *reinterpret_cast<const uint4*>(o);
}

// ======== pre-pass: V f32 [b,s,h,e] -> bf16 transposed [b,h,e,s] ========
__global__ __launch_bounds__(256)
void cvt_v_tr_bf16(const float* __restrict__ V, __bf16* __restrict__ Vt)
{
    __shared__ __bf16 tile[32][E_ + 8];
    const int tid = threadIdx.x;
    const int s0 = blockIdx.x * 32;
    const int h  = blockIdx.y;
    const int b  = blockIdx.z;
    {
        const int s  = tid >> 3;
        const int e0 = (tid & 7) * 8;
        const float* src = V + (((size_t)b * L_ + s0 + s) * H_ + h) * E_ + e0;
        const float4 a = *reinterpret_cast<const float4*>(src);
        const float4 c = *reinterpret_cast<const float4*>(src + 4);
        __align__(16) __bf16 o[8] = { (__bf16)a.x,(__bf16)a.y,(__bf16)a.z,(__bf16)a.w,
                                      (__bf16)c.x,(__bf16)c.y,(__bf16)c.z,(__bf16)c.w };
        *reinterpret_cast<uint4*>(&tile[s][e0]) = *reinterpret_cast<const uint4*>(o);
    }
    __syncthreads();
    {
        const int e  = tid >> 2;
        const int sc = (tid & 3) * 8;
        __align__(16) __bf16 o[8];
#pragma unroll
        for (int i = 0; i < 8; ++i) o[i] = tile[sc + i][e];
        __bf16* dst = Vt + (((size_t)b * H_ + h) * E_ + e) * L_ + s0 + sc;
        *reinterpret_cast<uint4*>(dst) = *reinterpret_cast<const uint4*>(o);
    }
}

// ================= main kernel: bf16 K/V^T from workspace =================
__global__ __launch_bounds__(256)
void fa_fwd_bf16(const float*  __restrict__ Q,
                 const __bf16* __restrict__ Kb,
                 const __bf16* __restrict__ Vtb,
                 float* __restrict__ O)
{
    __shared__ __bf16 Ks[2][BN][KBSTR];     // K tile  [key][e], double-buffered
    __shared__ __bf16 Vs[2][E_][VBSTR];     // V^T tile [e][key]
    __shared__ __bf16 Ps[NWAVE][BN][PSTR];  // P^T staging [key][qrow]

    const int tid  = threadIdx.x;
    const int wave = tid >> 5;
    const int lane = tid & 31;
    const int half = lane >> 4;
    const int ln   = lane & 15;

    const int b  = blockIdx.z;
    const int h  = blockIdx.y;
    const int m0 = blockIdx.x * BM;

    const float scale = 0.125f;

    const __bf16* kbase = Kb  + ((size_t)b * H_ + h) * L_ * E_;
    const __bf16* vbase = Vtb + ((size_t)b * H_ + h) * E_ * L_;

    // async staging coords: one b128 (8 bf16) per tensor per thread
    const int kkey = tid >> 3, kch = tid & 7;   // K: 32 keys x 8 chunks
    const int ve   = tid >> 2, vch = tid & 3;   // V^T: 64 e-rows x 4 chunks

    auto issue = [&](int kt, int nb) {
        const unsigned lk = (unsigned)(uintptr_t)&Ks[nb][kkey][kch * 8];
        const unsigned gk = (unsigned)((((kt * BN + kkey) * E_) + kch * 8) * 2);
        asm volatile("global_load_async_to_lds_b128 %0, %1, %2"
                     :: "v"(lk), "v"(gk),
                        "s"((unsigned long long)(uintptr_t)kbase) : "memory");
        const unsigned lv = (unsigned)(uintptr_t)&Vs[nb][ve][vch * 8];
        const unsigned gv = (unsigned)((ve * L_ + kt * BN + vch * 8) * 2);
        asm volatile("global_load_async_to_lds_b128 %0, %1, %2"
                     :: "v"(lv), "v"(gv),
                        "s"((unsigned long long)(uintptr_t)vbase) : "memory");
    };

    // ---- Q tile -> bf16 A-fragments (pre-scaled) ----
    const int qrow = m0 + wave * 16 + ln;
    const float* qp = Q + (((size_t)b * L_ + qrow) * H_ + h) * E_;
    v16bf qA[2];
#pragma unroll
    for (int c = 0; c < 2; ++c) {
        const int base = c * 32 + half * 8;
        const float4 f0 = *reinterpret_cast<const float4*>(qp + base);
        const float4 f1 = *reinterpret_cast<const float4*>(qp + base + 4);
        const float4 f2 = *reinterpret_cast<const float4*>(qp + base + 16);
        const float4 f3 = *reinterpret_cast<const float4*>(qp + base + 20);
        const float tf[16] = { f0.x,f0.y,f0.z,f0.w, f1.x,f1.y,f1.z,f1.w,
                               f2.x,f2.y,f2.z,f2.w, f3.x,f3.y,f3.z,f3.w };
#pragma unroll
        for (int i = 0; i < 16; ++i) qA[c][i] = (__bf16)(tf[i] * scale);
    }

    // constant all-ones B fragment (softmax denominator on the XDL pipe)
    v16bf onesb;
#pragma unroll
    for (int i = 0; i < 16; ++i) onesb[i] = (__bf16)1.0f;

    float mrow[8];
    v8f oacc[4], lacc;
#pragma unroll
    for (int j = 0; j < 8; ++j) { mrow[j] = -INFINITY; lacc[j] = 0.0f; }
#pragma unroll
    for (int t = 0; t < 4; ++t)
#pragma unroll
        for (int j = 0; j < 8; ++j) oacc[t][j] = 0.0f;

    const int NT = L_ / BN;
    issue(0, 0);

    for (int kt = 0; kt < NT; ++kt) {
        const int cb = kt & 1;
        if (kt + 1 < NT) {
            issue(kt + 1, cb ^ 1);
            asm volatile("s_wait_asynccnt 0x2" ::: "memory");
        } else {
            asm volatile("s_wait_asynccnt 0x0" ::: "memory");
        }
        __syncthreads();

        // ---- preload all K fragments, then 4 score WMMAs ----
        v16bf kbf[4];
#pragma unroll
        for (int c = 0; c < 2; ++c)
#pragma unroll
            for (int t = 0; t < 2; ++t) {
                const __bf16* src = &Ks[cb][t * 16 + ln][c * 32 + half * 8];
#pragma unroll
                for (int i = 0; i < 16; ++i) kbf[c * 2 + t][i] = src[i < 8 ? i : i + 8];
            }

        v8f sacc[2];
#pragma unroll
        for (int t = 0; t < 2; ++t)
#pragma unroll
            for (int j = 0; j < 8; ++j) sacc[t][j] = 0.0f;
#pragma unroll
        for (int c = 0; c < 2; ++c)
#pragma unroll
            for (int t = 0; t < 2; ++t)
                sacc[t] = __builtin_amdgcn_wmma_f32_16x16x32_bf16(
                    false, qA[c], false, kbf[c * 2 + t], (short)0, sacc[t], false, false);

        // ---- issue V fragment loads early: LDS latency hides under softmax VALU ----
        v16bf vbf[4];
#pragma unroll
        for (int t = 0; t < 4; ++t) {
            const __bf16* vsrc = &Vs[cb][t * 16 + ln][half * 16];
#pragma unroll
            for (int i = 0; i < 16; ++i) vbf[t][i] = vsrc[i];
        }

        // ---- online softmax: row max in VALU, row sum via ones-WMMA ----
        float p0a[8], p1a[8];
#pragma unroll
        for (int j = 0; j < 8; ++j) {
            const float mx    = redmax16(fmax_raw(sacc[0][j], sacc[1][j]));
            const float mnew  = fmax_raw(mrow[j], mx);
            const float alpha = __expf(mrow[j] - mnew);
            mrow[j] = mnew;
            p0a[j] = __expf(sacc[0][j] - mnew);
            p1a[j] = __expf(sacc[1][j] - mnew);
            lacc[j] *= alpha;
#pragma unroll
            for (int t = 0; t < 4; ++t) oacc[t][j] *= alpha;
        }

        // ---- pack P^T into LDS ----
        {
            __align__(16) __bf16 t0[8], t1[8];
#pragma unroll
            for (int j = 0; j < 8; ++j) { t0[j] = (__bf16)p0a[j]; t1[j] = (__bf16)p1a[j]; }
            *reinterpret_cast<uint4*>(&Ps[wave][ln][half * 8]) =
                *reinterpret_cast<const uint4*>(t0);
            *reinterpret_cast<uint4*>(&Ps[wave][16 + ln][half * 8]) =
                *reinterpret_cast<const uint4*>(t1);
        }

        // ---- read back P in A-layout ----
        v16bf pa;
#pragma unroll
        for (int i = 0; i < 16; ++i) {
            const int kv = half * 16 + i;
            pa[i] = Ps[wave][kv][ln];
        }

        // ---- O += P.V ; l += P.1 ----
#pragma unroll
        for (int t = 0; t < 4; ++t)
            oacc[t] = __builtin_amdgcn_wmma_f32_16x16x32_bf16(
                false, pa, false, vbf[t], (short)0, oacc[t], false, false);
        lacc = __builtin_amdgcn_wmma_f32_16x16x32_bf16(
            false, pa, false, onesb, (short)0, lacc, false, false);

        __syncthreads();
    }

    // ---- epilogue ----
#pragma unroll
    for (int j = 0; j < 8; ++j) {
        const float inv = 1.0f / lacc[j];
        const int row = m0 + wave * 16 + half * 8 + j;
        float* op = O + (((size_t)b * L_ + row) * H_ + h) * E_;
#pragma unroll
        for (int t = 0; t < 4; ++t) op[t * 16 + ln] = oacc[t][j] * inv;
    }
}

// ============ fallback (workspace too small): self-contained f32-async kernel ============
__global__ __launch_bounds__(256)
void fa_fwd_f32(const float* __restrict__ Q,
                const float* __restrict__ K,
                const float* __restrict__ V,
                float* __restrict__ O)
{
    __shared__ float  Kf[2][BN][KSTR];
    __shared__ float  Vt[2][E_][VSTR];
    __shared__ __bf16 Ps[NWAVE][BN][PSTR];

    const int tid  = threadIdx.x;
    const int wave = tid >> 5;
    const int lane = tid & 31;
    const int half = lane >> 4;
    const int ln   = lane & 15;
    const int b  = blockIdx.z;
    const int h  = blockIdx.y;
    const int m0 = blockIdx.x * BM;
    const float scale = 0.125f;

    const float* kbase = K + ((size_t)b * L_ * H_ + h) * E_;
    const float* vbase = V + ((size_t)b * L_ * H_ + h) * E_;
    const int skey = tid >> 3;
    const int se0  = (tid & 7) * 8;

    auto issue = [&](int kt, int nb) {
        const unsigned voff = (unsigned)(((kt * BN + skey) * (H_ * E_) + se0) * 4);
        const unsigned lk   = (unsigned)(uintptr_t)&Kf[nb][skey][se0];
        asm volatile("global_load_async_to_lds_b128 %0, %1, %2"
                     :: "v"(lk), "v"(voff),
                        "s"((unsigned long long)(uintptr_t)kbase) : "memory");
        asm volatile("global_load_async_to_lds_b128 %0, %1, %2 offset:16"
                     :: "v"(lk), "v"(voff),
                        "s"((unsigned long long)(uintptr_t)kbase) : "memory");
#pragma unroll
        for (int i = 0; i < 8; ++i) {
            const unsigned lv = (unsigned)(uintptr_t)&Vt[nb][se0 + i][skey];
            const unsigned go = voff + i * 4;
            asm volatile("global_load_async_to_lds_b32 %0, %1, %2"
                         :: "v"(lv), "v"(go),
                            "s"((unsigned long long)(uintptr_t)vbase) : "memory");
        }
    };

    const int qrow = m0 + wave * 16 + ln;
    const float* qp = Q + (((size_t)b * L_ + qrow) * H_ + h) * E_;
    v16bf qA[2];
#pragma unroll
    for (int c = 0; c < 2; ++c) {
        const int base = c * 32 + half * 8;
        const float4 f0 = *reinterpret_cast<const float4*>(qp + base);
        const float4 f1 = *reinterpret_cast<const float4*>(qp + base + 4);
        const float4 f2 = *reinterpret_cast<const float4*>(qp + base + 16);
        const float4 f3 = *reinterpret_cast<const float4*>(qp + base + 20);
        const float tf[16] = { f0.x,f0.y,f0.z,f0.w, f1.x,f1.y,f1.z,f1.w,
                               f2.x,f2.y,f2.z,f2.w, f3.x,f3.y,f3.z,f3.w };
#pragma unroll
        for (int i = 0; i < 16; ++i) qA[c][i] = (__bf16)(tf[i] * scale);
    }

    float mrow[8], lrow[8];
    v8f oacc[4];
#pragma unroll
    for (int j = 0; j < 8; ++j) { mrow[j] = -INFINITY; lrow[j] = 0.0f; }
#pragma unroll
    for (int t = 0; t < 4; ++t)
#pragma unroll
        for (int j = 0; j < 8; ++j) oacc[t][j] = 0.0f;

    const int NT = L_ / BN;
    issue(0, 0);

    for (int kt = 0; kt < NT; ++kt) {
        const int cb = kt & 1;
        if (kt + 1 < NT) {
            issue(kt + 1, cb ^ 1);
            asm volatile("s_wait_asynccnt 0xa" ::: "memory");
        } else {
            asm volatile("s_wait_asynccnt 0x0" ::: "memory");
        }
        __syncthreads();

        v8f sacc[2];
#pragma unroll
        for (int t = 0; t < 2; ++t)
#pragma unroll
            for (int j = 0; j < 8; ++j) sacc[t][j] = 0.0f;

#pragma unroll
        for (int c = 0; c < 2; ++c)
#pragma unroll
            for (int t = 0; t < 2; ++t) {
                const float* src = &Kf[cb][t * 16 + ln][c * 32 + half * 8];
                v16bf kb;
#pragma unroll
                for (int i = 0; i < 16; ++i) kb[i] = (__bf16)src[i < 8 ? i : i + 8];
                sacc[t] = __builtin_amdgcn_wmma_f32_16x16x32_bf16(
                    false, qA[c], false, kb, (short)0, sacc[t], false, false);
            }

        float p0a[8], p1a[8];
#pragma unroll
        for (int j = 0; j < 8; ++j) {
            const float mx    = redmax16(fmax_raw(sacc[0][j], sacc[1][j]));
            const float mnew  = fmax_raw(mrow[j], mx);
            const float alpha = __expf(mrow[j] - mnew);
            mrow[j] = mnew;
            const float p0 = __expf(sacc[0][j] - mnew);
            const float p1 = __expf(sacc[1][j] - mnew);
            const float rs = redsum16(p0 + p1);
            lrow[j] = lrow[j] * alpha + rs;
            p0a[j] = p0; p1a[j] = p1;
#pragma unroll
            for (int t = 0; t < 4; ++t) oacc[t][j] *= alpha;
        }

        {
            __align__(16) __bf16 t0[8], t1[8];
#pragma unroll
            for (int j = 0; j < 8; ++j) { t0[j] = (__bf16)p0a[j]; t1[j] = (__bf16)p1a[j]; }
            *reinterpret_cast<uint4*>(&Ps[wave][ln][half * 8]) =
                *reinterpret_cast<const uint4*>(t0);
            *reinterpret_cast<uint4*>(&Ps[wave][16 + ln][half * 8]) =
                *reinterpret_cast<const uint4*>(t1);
        }

        v16bf pa;
#pragma unroll
        for (int i = 0; i < 16; ++i) {
            const int kv = half * 16 + i;
            pa[i] = Ps[wave][kv][ln];
        }

#pragma unroll
        for (int t = 0; t < 4; ++t) {
            const float* vsrc = &Vt[cb][t * 16 + ln][half * 16];
            v16bf vb;
#pragma unroll
            for (int i = 0; i < 16; ++i) vb[i] = (__bf16)vsrc[i];
            oacc[t] = __builtin_amdgcn_wmma_f32_16x16x32_bf16(
                false, pa, false, vb, (short)0, oacc[t], false, false);
        }
        __syncthreads();
    }

#pragma unroll
    for (int j = 0; j < 8; ++j) {
        const float inv = 1.0f / lrow[j];
        const int row = m0 + wave * 16 + half * 8 + j;
        float* op = O + (((size_t)b * L_ + row) * H_ + h) * E_;
#pragma unroll
        for (int t = 0; t < 4; ++t) op[t * 16 + ln] = oacc[t][j] * inv;
    }
}

extern "C" void kernel_launch(void* const* d_in, const int* in_sizes, int n_in,
                              void* d_out, int out_size, void* d_ws, size_t ws_size,
                              hipStream_t stream) {
    (void)in_sizes; (void)n_in; (void)out_size;
    const float* Q = (const float*)d_in[0];
    const float* K = (const float*)d_in[1];
    const float* V = (const float*)d_in[2];
    float* O = (float*)d_out;

    const size_t elems = (size_t)B_ * H_ * L_ * E_;
    const size_t need  = 2 * elems * sizeof(__bf16);
    dim3 grid(L_ / BM, H_, B_);
    dim3 block(256);

    if (d_ws != nullptr && ws_size >= need) {
        __bf16* Kb  = (__bf16*)d_ws;
        __bf16* Vtb = Kb + elems;
        const int nblk = (int)(elems / 8 / 256);
        hipLaunchKernelGGL(cvt_k_bf16, dim3(nblk), block, 0, stream, K, Kb);
        dim3 gv(L_ / 32, H_, B_);
        hipLaunchKernelGGL(cvt_v_tr_bf16, gv, block, 0, stream, V, Vtb);
        hipLaunchKernelGGL(fa_fwd_bf16, grid, block, 0, stream, Q, Kb, Vtb, O);
    } else {
        hipLaunchKernelGGL(fa_fwd_f32, grid, block, 0, stream, Q, K, V, O);
    }
}
